// LateJoinGConv_13228499272261
// MI455X (gfx1250) — matrix operand order repaired and checked by reading, not compiled, not run on hardware
//
#include <hip/hip_runtime.h>
#include <hip/hip_bf16.h>

typedef __attribute__((ext_vector_type(16))) _Float16 v16h;
typedef __attribute__((ext_vector_type(8)))  _Float16 v8h;
typedef __attribute__((ext_vector_type(8)))  float    v8f;

#define BN_NODES 256000
#define BM_EDGES 512000
#define B_GRAPHS 64
#define NODES_PER_G (BN_NODES / B_GRAPHS)   // 4000
#define CPG_CONF 1000
#define OUT_N (B_GRAPHS * CPG_CONF)         // 64000
#define HDIM 64
#define K0 151     // GNN input dim (139 + 8 + 4)
#define K0PAD 160
#define KM 152     // MLP input dim (128 + 24)
#define KMPAD 160

// ---------------------------------------------------------------- utilities
__global__ void fill0_kernel(float* __restrict__ p, size_t n) {
    size_t i = (size_t)blockIdx.x * blockDim.x + threadIdx.x;
    size_t stride = (size_t)gridDim.x * blockDim.x;
    for (; i < n; i += stride) p[i] = 0.0f;
}

// ---------------------------------------------------------------- embedding
// X0[n][0:139]=node_feat cont, [139:147]=op_emb, [147:151]=shape_emb, [151:160]=0
__global__ void embed_kernel(const float* __restrict__ nf,
                             const int* __restrict__ opc,
                             const float* __restrict__ op_emb,
                             const float* __restrict__ shape_emb,
                             float* __restrict__ X0) {
    size_t idx = (size_t)blockIdx.x * blockDim.x + threadIdx.x;
    if (idx >= (size_t)BN_NODES * K0PAD) return;
    int n = (int)(idx / K0PAD);
    int c = (int)(idx % K0PAD);
    float v;
    if (c < 139) {
        v = nf[(size_t)n * 140 + c];
    } else if (c < 147) {
        v = op_emb[opc[n] * 8 + (c - 139)];
    } else if (c < 151) {
        int sid = (int)nf[(size_t)n * 140 + 139];
        sid = sid < 0 ? 0 : (sid > 7 ? 7 : sid);
        v = shape_emb[sid * 4 + (c - 147)];
    } else {
        v = 0.0f;
    }
    X0[idx] = v;
}

// ---------------------------------------------------------------- weight pack
// Pack fp32 W (Krows x 64) into fp16 B-operand lane layout (Kpad x 64, zero pad):
// pack index = ((kblk*4 + nb)*32 + lane)*16 + j
//   lane<16: K = kblk*32 + (j<8 ? j : j+8),     N = nb*16 + lane
//   lane>=16: K = kblk*32 + 8 + (j<8 ? j : j+8), N = nb*16 + (lane-16)
__global__ void pack_w_kernel(const float* __restrict__ W,
                              _Float16* __restrict__ P,
                              int Krows, int Kpad) {
    int idx = blockIdx.x * blockDim.x + threadIdx.x;
    int total = Kpad * 64;
    if (idx >= total) return;
    int j    = idx & 15;
    int lane = (idx >> 4) & 31;
    int nb   = (idx >> 9) & 3;
    int kblk = idx >> 11;
    int krow = kblk * 32 + ((lane < 16) ? 0 : 8) + ((j < 8) ? j : j + 8);
    int col  = nb * 16 + (lane & 15);
    float v = (krow < Krows) ? W[krow * 64 + col] : 0.0f;
    P[idx] = (_Float16)v;
}

// ---------------------------------------------------------------- graph ops
__global__ void degree_kernel(const int* __restrict__ edges,
                              float* __restrict__ deg) {
    int e = blockIdx.x * blockDim.x + threadIdx.x;
    if (e >= BM_EDGES) return;
    atomicAdd(&deg[edges[e * 2 + 0]], 1.0f);
    atomicAdd(&deg[edges[e * 2 + 1]], 1.0f);
}

// one wave per edge; lanes cover channels (coalesced)
__global__ void aggregate_kernel(const int* __restrict__ edges,
                                 const float* __restrict__ X,
                                 float* __restrict__ MSG,
                                 int ld, int dim) {
    int t = blockIdx.x * blockDim.x + threadIdx.x;
    int e = t >> 5;
    int lane = t & 31;
    if (e >= BM_EDGES) return;
    int s = edges[e * 2 + 0];
    int d = edges[e * 2 + 1];
    const float* xs = X + (size_t)s * ld;
    const float* xd = X + (size_t)d * ld;
    float* ms = MSG + (size_t)s * ld;
    float* md = MSG + (size_t)d * ld;
    for (int c = lane; c < dim; c += 32) {
        atomicAdd(&md[c], xs[c]);
        atomicAdd(&ms[c], xd[c]);
    }
}

// ---------------------------------------------------------------- WMMA GEMM
// A-operand f16 16x32 lane layout (ISA 7.12.2): lane holds row m0+(lane&15);
// j=0..7 -> K = k + d, j=8..15 -> K = k+16 + d,   d = (lane<16 ? 0 : 8)
__device__ __forceinline__ v16h load_a16(const float* __restrict__ rowp,
                                         int k, int lane, float scale) {
    const float* p0 = rowp + k + ((lane & 16) ? 8 : 0);
    const float* p1 = p0 + 16;
    v16h a;
#pragma unroll
    for (int j = 0; j < 8; ++j) a[j] = (_Float16)(p0[j] * scale);
#pragma unroll
    for (int j = 0; j < 8; ++j) a[8 + j] = (_Float16)(p1[j] * scale);
    return a;
}

// Y(ntiles*16 x 64) = relu( A0 @ B0 + (A1/deg) @ B1 + bias ), Kpad x 64 weights.
// One wave computes a 16x64 output tile via 4 WMMA accumulators. Packed weight
// operands are staged once per workgroup into LDS (shared by all 8 waves),
// turning the hot-loop B reads into ds_loads and cutting L2 traffic 8x.
__global__ void wmma_gemm_kernel(const float* __restrict__ A0,
                                 const _Float16* __restrict__ B0,
                                 const float* __restrict__ A1,
                                 const _Float16* __restrict__ B1,
                                 const float* __restrict__ deg,
                                 const float* __restrict__ bias,
                                 float* __restrict__ Y,
                                 int ld, int kpad, int ntiles) {
    __shared__ alignas(32) _Float16 Bs0[K0PAD * 64];
    __shared__ alignas(32) _Float16 Bs1[K0PAD * 64];

    const int tid = threadIdx.x;
    const int total = kpad * 64;  // packed halves per matrix
    for (int i = tid * 8; i < total; i += blockDim.x * 8)
        *(v8h*)(Bs0 + i) = *(const v8h*)(B0 + i);
    if (A1 != nullptr) {
        for (int i = tid * 8; i < total; i += blockDim.x * 8)
            *(v8h*)(Bs1 + i) = *(const v8h*)(B1 + i);
    }
    __syncthreads();

    int wid = (int)((blockIdx.x * blockDim.x + tid) >> 5);
    int lane = tid & 31;
    if (wid >= ntiles) return;  // wave-uniform: EXEC stays all-ones for WMMA
    int m0 = wid * 16;
    int r = m0 + (lane & 15);

    v8f acc[4] = {};

    const float* a0row = A0 + (size_t)r * ld;
    for (int k = 0; k < kpad; k += 32) {
        __builtin_prefetch(a0row + k + 32, 0, 0);  // global_prefetch_b8
        v16h a = load_a16(a0row, k, lane, 1.0f);
        const _Float16* bp = Bs0 + (k >> 5) * 2048 + lane * 16;
#pragma unroll
        for (int nb = 0; nb < 4; ++nb) {
            v16h b = *(const v16h*)(bp + nb * 512);
            acc[nb] = __builtin_amdgcn_wmma_f32_16x16x32_f16(
                false, a, false, b, (short)0, acc[nb], false, false);
        }
    }
    if (A1 != nullptr) {
        const float* a1row = A1 + (size_t)r * ld;
        float inv = 1.0f / fmaxf(deg[r], 1.0f);
        for (int k = 0; k < kpad; k += 32) {
            __builtin_prefetch(a1row + k + 32, 0, 0);
            v16h a = load_a16(a1row, k, lane, inv);
            const _Float16* bp = Bs1 + (k >> 5) * 2048 + lane * 16;
#pragma unroll
            for (int nb = 0; nb < 4; ++nb) {
                v16h b = *(const v16h*)(bp + nb * 512);
                acc[nb] = __builtin_amdgcn_wmma_f32_16x16x32_f16(
                    false, a, false, b, (short)0, acc[nb], false, false);
            }
        }
    }

    // C/D layout: lane<16 -> rows m0+i, lane>=16 -> rows m0+8+i ; col = lane&15
    int mbase = m0 + ((lane & 16) ? 8 : 0);
    int ncol = lane & 15;
#pragma unroll
    for (int nb = 0; nb < 4; ++nb) {
        int n = nb * 16 + ncol;
        float bv = bias[n];
#pragma unroll
        for (int i = 0; i < 8; ++i) {
            float v = acc[nb][i] + bv;
            v = fmaxf(v, 0.0f);
            Y[(size_t)(mbase + i) * 64 + n] = v;
        }
    }
}

// ---------------------------------------------------------------- pooling
// batch = node/4000 is contiguous: block = graph, thread = channel
__global__ void pool_kernel(const float* __restrict__ X, float* __restrict__ XG) {
    int g = blockIdx.x;
    int c = threadIdx.x;  // 0..63
    const float* p = X + (size_t)g * NODES_PER_G * 64 + c;
    float sum = 0.0f, mx = -3.402823466e38f;
    for (int n = 0; n < NODES_PER_G; ++n) {
        float v = p[(size_t)n * 64];
        sum += v;
        mx = fmaxf(mx, v);
    }
    XG[g * 128 + c] = sum * (1.0f / (float)NODES_PER_G);
    XG[g * 128 + 64 + c] = mx;
}

// XC[r][0:128]=XG[r/1000], [128:152]=config_feat[r], [152:160]=0
__global__ void build_xc_kernel(const float* __restrict__ XG,
                                const float* __restrict__ cfg,
                                float* __restrict__ XC) {
    size_t idx = (size_t)blockIdx.x * blockDim.x + threadIdx.x;
    if (idx >= (size_t)OUT_N * KMPAD) return;
    int r = (int)(idx / KMPAD);
    int c = (int)(idx % KMPAD);
    int g = r / CPG_CONF;
    float v;
    if (c < 128)       v = XG[g * 128 + c];
    else if (c < 152)  v = cfg[(size_t)r * 24 + (c - 128)];
    else               v = 0.0f;
    XC[idx] = v;
}

__global__ void final_kernel(const float* __restrict__ H,
                             const float* __restrict__ W2,
                             const float* __restrict__ b2,
                             float* __restrict__ out) {
    int r = blockIdx.x * blockDim.x + threadIdx.x;
    if (r >= OUT_N) return;
    const float* h = H + (size_t)r * 64;
    float s = 0.0f;
#pragma unroll
    for (int i = 0; i < 64; ++i) s += h[i] * W2[i];
    out[r] = s + b2[0];
}

// ---------------------------------------------------------------- launch
extern "C" void kernel_launch(void* const* d_in, const int* in_sizes, int n_in,
                              void* d_out, int out_size, void* d_ws, size_t ws_size,
                              hipStream_t stream) {
    const float* node_feat   = (const float*)d_in[0];
    const int*   node_opcode = (const int*)d_in[1];
    const int*   edge_index  = (const int*)d_in[2];
    // d_in[3] batch (contiguous: node/4000), d_in[4] n_configs (all 1000) unused
    const float* config_feat = (const float*)d_in[5];
    const float* op_emb      = (const float*)d_in[6];
    const float* shape_emb   = (const float*)d_in[7];
    const float* Ws_self[3]  = {(const float*)d_in[8],  (const float*)d_in[9],  (const float*)d_in[10]};
    const float* Ws_nbr[3]   = {(const float*)d_in[11], (const float*)d_in[12], (const float*)d_in[13]};
    const float* bs[3]       = {(const float*)d_in[14], (const float*)d_in[15], (const float*)d_in[16]};
    const float* post_W1     = (const float*)d_in[17];
    const float* post_b1     = (const float*)d_in[18];
    const float* post_W2     = (const float*)d_in[19];
    const float* post_b2     = (const float*)d_in[20];
    float* out = (float*)d_out;

    char* ws = (char*)d_ws;
    size_t off = 0;
    auto alloc = [&](size_t bytes) -> void* {
        void* p = ws + off;
        off += (bytes + 255) & ~(size_t)255;
        return p;
    };
    float* X0   = (float*)alloc((size_t)BN_NODES * K0PAD * 4);
    float* X1   = (float*)alloc((size_t)BN_NODES * 64 * 4);
    float* X2   = (float*)alloc((size_t)BN_NODES * 64 * 4);
    float* MSG  = (float*)alloc((size_t)BN_NODES * K0PAD * 4);
    float* DEG  = (float*)alloc((size_t)BN_NODES * 4);
    float* XG   = (float*)alloc((size_t)B_GRAPHS * 128 * 4);
    _Float16* Wp_self[3];
    _Float16* Wp_nbr[3];
    Wp_self[0] = (_Float16*)alloc((size_t)K0PAD * 64 * 2);
    Wp_nbr[0]  = (_Float16*)alloc((size_t)K0PAD * 64 * 2);
    Wp_self[1] = (_Float16*)alloc((size_t)64 * 64 * 2);
    Wp_nbr[1]  = (_Float16*)alloc((size_t)64 * 64 * 2);
    Wp_self[2] = (_Float16*)alloc((size_t)64 * 64 * 2);
    Wp_nbr[2]  = (_Float16*)alloc((size_t)64 * 64 * 2);
    _Float16* Wp_post = (_Float16*)alloc((size_t)KMPAD * 64 * 2);
    // reuse big buffers that are dead by the time these are needed
    float* XC = X0;    // 64000*160 floats << X0
    float* H  = MSG;   // 64000*64  floats << MSG

    // --- embed + weight packing ---
    {
        size_t tot = (size_t)BN_NODES * K0PAD;
        embed_kernel<<<(unsigned)((tot + 255) / 256), 256, 0, stream>>>(
            node_feat, node_opcode, op_emb, shape_emb, X0);
    }
    pack_w_kernel<<<(K0PAD * 64 + 255) / 256, 256, 0, stream>>>(Ws_self[0], Wp_self[0], K0, K0PAD);
    pack_w_kernel<<<(K0PAD * 64 + 255) / 256, 256, 0, stream>>>(Ws_nbr[0],  Wp_nbr[0],  K0, K0PAD);
    pack_w_kernel<<<(64 * 64 + 255) / 256, 256, 0, stream>>>(Ws_self[1], Wp_self[1], 64, 64);
    pack_w_kernel<<<(64 * 64 + 255) / 256, 256, 0, stream>>>(Ws_nbr[1],  Wp_nbr[1],  64, 64);
    pack_w_kernel<<<(64 * 64 + 255) / 256, 256, 0, stream>>>(Ws_self[2], Wp_self[2], 64, 64);
    pack_w_kernel<<<(64 * 64 + 255) / 256, 256, 0, stream>>>(Ws_nbr[2],  Wp_nbr[2],  64, 64);
    pack_w_kernel<<<(KMPAD * 64 + 255) / 256, 256, 0, stream>>>(post_W1, Wp_post, KM, KMPAD);

    // --- degrees (symmetric, constant across layers) ---
    fill0_kernel<<<1024, 256, 0, stream>>>(DEG, (size_t)BN_NODES);
    degree_kernel<<<(BM_EDGES + 255) / 256, 256, 0, stream>>>(edge_index, DEG);

    const int agg_blocks = (BM_EDGES * 32) / 256;  // wave per edge
    const int sage_blocks = (BN_NODES / 16) / 8;   // 8 waves/block, 16 rows/wave

    // --- layer 0: K=151 (pad 160) ---
    fill0_kernel<<<4096, 256, 0, stream>>>(MSG, (size_t)BN_NODES * K0PAD);
    aggregate_kernel<<<agg_blocks, 256, 0, stream>>>(edge_index, X0, MSG, K0PAD, K0);
    wmma_gemm_kernel<<<sage_blocks, 256, 0, stream>>>(
        X0, Wp_self[0], MSG, Wp_nbr[0], DEG, bs[0], X1, K0PAD, K0PAD, BN_NODES / 16);

    // --- layer 1: K=64 ---
    fill0_kernel<<<4096, 256, 0, stream>>>(MSG, (size_t)BN_NODES * 64);
    aggregate_kernel<<<agg_blocks, 256, 0, stream>>>(edge_index, X1, MSG, 64, 64);
    wmma_gemm_kernel<<<sage_blocks, 256, 0, stream>>>(
        X1, Wp_self[1], MSG, Wp_nbr[1], DEG, bs[1], X2, 64, 64, BN_NODES / 16);

    // --- layer 2: K=64 (output back into X1) ---
    fill0_kernel<<<4096, 256, 0, stream>>>(MSG, (size_t)BN_NODES * 64);
    aggregate_kernel<<<agg_blocks, 256, 0, stream>>>(edge_index, X2, MSG, 64, 64);
    wmma_gemm_kernel<<<sage_blocks, 256, 0, stream>>>(
        X2, Wp_self[2], MSG, Wp_nbr[2], DEG, bs[2], X1, 64, 64, BN_NODES / 16);

    // --- pooling (avg + max per graph) ---
    pool_kernel<<<B_GRAPHS, 64, 0, stream>>>(X1, XG);

    // --- build per-config features and MLP head ---
    {
        size_t tot = (size_t)OUT_N * KMPAD;
        build_xc_kernel<<<(unsigned)((tot + 255) / 256), 256, 0, stream>>>(XG, config_feat, XC);
    }
    wmma_gemm_kernel<<<(OUT_N / 16) / 8, 256, 0, stream>>>(
        XC, Wp_post, nullptr, nullptr, DEG, post_b1, H, KMPAD, KMPAD, OUT_N / 16);
    final_kernel<<<(OUT_N + 255) / 256, 256, 0, stream>>>(H, post_W2, post_b2, out);
}
